// DHS_57784490001220
// MI455X (gfx1250) — matrix operand order repaired and checked by reading, not compiled, not run on hardware
//
#include <hip/hip_runtime.h>
#include <hip/hip_bf16.h>

typedef __attribute__((ext_vector_type(16))) _Float16 v16h;
typedef __attribute__((ext_vector_type(8)))  _Float16 v8h;
typedef __attribute__((ext_vector_type(4)))  _Float16 v4h;
typedef __attribute__((ext_vector_type(8)))  float    v8f;

#define BB   256
#define LL   75
#define UU   64
#define TS   16      // j-tile rows per workgroup
#define KW   32      // k-window rows staged in LDS ([j0-8, j0+24))
#define STR  68      // f32 LDS row stride: 272B -> 16B-aligned rows, 4-bank rotation
#define STRH 72      // f16 LDS row stride (halfs): 144B -> 16B-aligned rows

__device__ __forceinline__ float fast_tanh(float x) {
#if __has_builtin(__builtin_amdgcn_tanhf)
    return __builtin_amdgcn_tanhf(x);          // v_tanh_f32
#elif __has_builtin(__builtin_amdgcn_tanh_f32)
    return __builtin_amdgcn_tanh_f32(x);
#else
    return tanhf(x);
#endif
}

// Fused windowed additive attention:
//   q = X*Wt^T (+bh), k = X*Wx^T   via v_wmma_f32_16x16x32_f16
//   e[j,k] = Wa . tanh(q_j + k_k + rel_{j-k+L-1}) + ba   only for k in [j-8, j+8)
//   (out-of-window logits underflow to exactly 0 after softmax in fp32)
__global__ __launch_bounds__(128) void fused_winattn(
    const float* __restrict__ X,    // [B,L,U]
    const float* __restrict__ Wt,   // [U,U]
    const float* __restrict__ Wx,   // [U,U]
    const float* __restrict__ bh,   // [U]
    const float* __restrict__ Wa,   // [1,U]
    const float* __restrict__ ba,   // [1]
    const float* __restrict__ rel,  // [2L-1,U]
    float* __restrict__ out)        // [B,L,U]
{
    __shared__ float    sX [KW * STR];   // X window rows (fp32, for AV)
    __shared__ _Float16 sXh[KW * STRH];  // X window rows (f16, WMMA A operands)
    __shared__ float    sQ [TS * STR];   // Q tile (+bh folded in)
    __shared__ float    sK [KW * STR];   // K projection of window rows
    __shared__ float    sR [16 * STR];   // rel rows 67..82 (filled by ASYNC loads)
    __shared__ float    sWa[UU];
    __shared__ float    sE [TS * 16];    // logits -> softmax weights (in place)

    const int tid  = threadIdx.x;
    const int lane = tid & 31;
    const int wv   = tid >> 5;           // wave 0..3 (wave32)
    const int b    = blockIdx.y;
    const int j0   = blockIdx.x * TS;

    const int n     = lane & 15;
    const int kbase = (lane >> 4) * 16;  // B operand: K base per half-wave
    const int koff  = (lane >> 4) * 8;   // unused: the A operand also uses kbase
    const int vcol  = wv * 16 + n;

    // Warm caches for this lane's weight rows (global_prefetch_b8).
    __builtin_prefetch(Wt + vcol * UU, 0, 0);
    __builtin_prefetch(Wx + vcol * UU, 0, 0);

    // ---- rel rows 67..82 -> LDS via async global->LDS DMA (ASYNCcnt path) ----
    // 16 rows x 16 float4 = 256 b128 transfers; always in-bounds (67..82 < 149).
    #pragma unroll
    for (int s = 0; s < 2; ++s) {
        int i  = tid + 128 * s;          // [0,256)
        int d  = i >> 4, u4 = i & 15;
        unsigned lds_addr = (unsigned)(size_t)&sR[d * STR + u4 * 4];
        unsigned long long gaddr =
            (unsigned long long)(size_t)(rel + (67 + d) * UU + u4 * 4);
        asm volatile("global_load_async_to_lds_b128 %0, %1, off"
                     :: "v"(lds_addr), "v"(gaddr) : "memory");
    }

    // ---- stage X window [j0-8, j0+24): fp32 copy + f16 copy (for WMMA A) ----
    const float4* Xg = (const float4*)X;
    #pragma unroll
    for (int s = 0; s < 4; ++s) {
        int i  = tid + 128 * s;          // [0, 512): 32 rows x 16 float4
        int t  = i >> 4, u4 = i & 15;
        int k  = j0 - 8 + t;
        float4 v = {0.f, 0.f, 0.f, 0.f};
        if (k >= 0 && k < LL) v = Xg[((long)b * LL + k) * 16 + u4];
        *(float4*)&sX[t * STR + u4 * 4] = v;
        v4h h;
        h[0] = (_Float16)v.x; h[1] = (_Float16)v.y;
        h[2] = (_Float16)v.z; h[3] = (_Float16)v.w;
        *(v4h*)&sXh[t * STRH + u4 * 4] = h;
    }
    if (tid < 16) ((float4*)sWa)[tid] = ((const float4*)Wa)[tid];

    asm volatile("s_wait_asynccnt 0x0" ::: "memory");   // rel tile landed
    __syncthreads();

    // ---- WMMA projections: wave wv owns output columns [16*wv, 16*wv+16) ----
    // B layout (32x16 KxN): lanes 0-15 -> K=0..15, lanes 16-31 -> K=16..31.
    v16h bWt[2], bWx[2];
    #pragma unroll
    for (int kp = 0; kp < 2; ++kp) {
        const float4* wtr = (const float4*)(Wt + vcol * UU + kp * 32 + kbase);
        const float4* wxr = (const float4*)(Wx + vcol * UU + kp * 32 + kbase);
        #pragma unroll
        for (int q = 0; q < 4; ++q) {
            float4 a = wtr[q], c = wxr[q];
            bWt[kp][q*4+0] = (_Float16)a.x; bWt[kp][q*4+1] = (_Float16)a.y;
            bWt[kp][q*4+2] = (_Float16)a.z; bWt[kp][q*4+3] = (_Float16)a.w;
            bWx[kp][q*4+0] = (_Float16)c.x; bWx[kp][q*4+1] = (_Float16)c.y;
            bWx[kp][q*4+2] = (_Float16)c.z; bWx[kp][q*4+3] = (_Float16)c.w;
        }
    }
    const float bh_v = bh[vcol];

    // A fragment: two aligned 16B ds_load_b128 per K=32 pass.
    // A layout (16x32 MxK f16): half-wave holds K=kbase+0..15.
    #define LOAD_A(row, kp, dst)                                                 \
        {                                                                        \
            const _Float16* xr_ = &sXh[(row) * STRH + (kp) * 32];                \
            v8h lo_ = *(const v8h*)&xr_[kbase];                                  \
            v8h hi_ = *(const v8h*)&xr_[kbase + 8];                              \
            dst = __builtin_shufflevector(lo_, hi_, 0,1,2,3,4,5,6,7,             \
                                                    8,9,10,11,12,13,14,15);      \
        }

    // Q tile: A rows are window rows 8..23 (== j rows j0..j0+15)
    {
        v8f acc = {0.f,0.f,0.f,0.f,0.f,0.f,0.f,0.f};
        #pragma unroll
        for (int kp = 0; kp < 2; ++kp) {
            v16h a; LOAD_A(8 + (lane & 15), kp, a);
            acc = __builtin_amdgcn_wmma_f32_16x16x32_f16(
                      false, a, false, bWt[kp], (short)0, acc, false, false);
        }
        #pragma unroll
        for (int r = 0; r < 8; ++r) {        // D layout: VGPR r -> M = r + 8*(lane>=16)
            int row = (lane >> 4) * 8 + r;
            sQ[row * STR + vcol] = acc[r] + bh_v;   // fold bh into Q
        }
    }
    // K projection: two 16-row M tiles covering all 32 window rows
    #pragma unroll
    for (int mt = 0; mt < 2; ++mt) {
        v8f acc = {0.f,0.f,0.f,0.f,0.f,0.f,0.f,0.f};
        #pragma unroll
        for (int kp = 0; kp < 2; ++kp) {
            v16h a; LOAD_A(mt * 16 + (lane & 15), kp, a);
            acc = __builtin_amdgcn_wmma_f32_16x16x32_f16(
                      false, a, false, bWx[kp], (short)0, acc, false, false);
        }
        #pragma unroll
        for (int r = 0; r < 8; ++r) {
            int row = mt * 16 + (lane >> 4) * 8 + r;
            sK[row * STR + vcol] = acc[r];
        }
    }
    __syncthreads();

    // ---- windowed emission: 16 rows x 16 window slots = 256 logits ----
    const float bav = ba[0];
    #pragma unroll
    for (int i = 0; i < 2; ++i) {
        int p = tid + 128 * i;               // [0,256)
        int r = p >> 4, w = p & 15;
        int t = r + w;                       // window-buffer row
        int j = j0 + r;
        int k = j - 8 + w;                   // global key index
        float e = -1e30f;
        if (j < LL && k >= 0 && k < LL) {
            const float4* qr  = (const float4*)&sQ[r * STR];
            const float4* kr  = (const float4*)&sK[t * STR];
            const float4* rr  = (const float4*)&sR[(15 - w) * STR]; // rel row 15-w
            const float4* wa4 = (const float4*)sWa;
            float s = 0.f;
            #pragma unroll 4
            for (int u4 = 0; u4 < 16; ++u4) {
                float4 q = qr[u4], kk = kr[u4], rv = rr[u4], w4 = wa4[u4];
                s += w4.x * fast_tanh(q.x + kk.x + rv.x);
                s += w4.y * fast_tanh(q.y + kk.y + rv.y);
                s += w4.z * fast_tanh(q.z + kk.z + rv.z);
                s += w4.w * fast_tanh(q.w + kk.w + rv.w);
            }
            e = s + bav;
        }
        sE[p] = e;
    }
    __syncthreads();

    // ---- softmax over each 16-wide window row ----
    if (tid < TS) {
        const int base = tid * 16;
        float m = -1e30f;
        #pragma unroll
        for (int w = 0; w < 16; ++w) m = fmaxf(m, sE[base + w]);
        float ex[16], sum = 0.f;
        #pragma unroll
        for (int w = 0; w < 16; ++w) { ex[w] = __expf(sE[base + w] - m); sum += ex[w]; }
        float inv = 1.f / sum;
        #pragma unroll
        for (int w = 0; w < 16; ++w) sE[base + w] = ex[w] * inv;
    }
    __syncthreads();

    // ---- v[j, u] = sum_w a[r][w] * X[window row r+w][u], float4-wide ----
    #pragma unroll
    for (int i = 0; i < 2; ++i) {
        int idx = tid + 128 * i;             // [0,256): 16 rows x 16 float4 cols
        int r = idx >> 4, u4 = idx & 15;
        int j = j0 + r;
        float4 acc = {0.f, 0.f, 0.f, 0.f};
        #pragma unroll
        for (int w = 0; w < 16; ++w) {
            float a  = sE[r * 16 + w];
            float4 x = *(const float4*)&sX[(r + w) * STR + u4 * 4];
            acc.x += a * x.x; acc.y += a * x.y;
            acc.z += a * x.z; acc.w += a * x.w;
        }
        if (j < LL)
            *(float4*)&out[((long)b * LL + j) * UU + u4 * 4] = acc;
    }
    #undef LOAD_A
}

extern "C" void kernel_launch(void* const* d_in, const int* in_sizes, int n_in,
                              void* d_out, int out_size, void* d_ws, size_t ws_size,
                              hipStream_t stream) {
    const float* X   = (const float*)d_in[0];   // inputs   [256,75,64]
    const float* Wt  = (const float*)d_in[1];   // Wt_w     [64,64]
    const float* Wx  = (const float*)d_in[2];   // Wx_w     [64,64]
    const float* bh  = (const float*)d_in[3];   // bh       [64]
    const float* Wa  = (const float*)d_in[4];   // Wa_w     [1,64]
    const float* ba  = (const float*)d_in[5];   // ba       [1]
    const float* rel = (const float*)d_in[6];   // rel_table[149,64]
    float* out = (float*)d_out;                 // [256,75,64]

    dim3 grid((LL + TS - 1) / TS, BB, 1);       // 5 x 256 workgroups
    dim3 block(128, 1, 1);                      // 4 wave32 waves
    fused_winattn<<<grid, block, 0, stream>>>(X, Wt, Wx, bh, Wa, ba, rel, out);
}
